// EnhancedContrastiveLoss_17841294147774
// MI455X (gfx1250) — matrix operand order repaired and checked by reading, not compiled, not run on hardware
//
#include <hip/hip_runtime.h>

#define BATCH 8192
#define DIM 256
#define INV_T 14.285714285714286f
#define MARGIN_V 0.2f
#define NEG_BIG -1e30f

typedef __attribute__((ext_vector_type(16))) _Float16 v16h;
typedef __attribute__((ext_vector_type(8)))  float    v8f;

// Load a 32-byte WMMA fragment from two 16-byte-aligned halves.
__device__ __forceinline__ v16h load_frag(const _Float16* p0, const _Float16* p1) {
    uint4 lo = *reinterpret_cast<const uint4*>(p0);
    uint4 hi = *reinterpret_cast<const uint4*>(p1);
    struct { uint4 a, b; } t{lo, hi};
    v16h r;
    __builtin_memcpy(&r, &t, sizeof(r));
    return r;
}

__global__ void zero_acc_kernel(float* acc) {
    if (threadIdx.x < 8) acc[threadIdx.x] = 0.0f;
}

// One wave (32 lanes) per row: L2-normalize fp32 row -> f16 row.
__global__ __launch_bounds__(256) void normalize_f16_kernel(
        const float* __restrict__ emb, _Float16* __restrict__ out) {
    int tid  = threadIdx.x;
    int w    = tid >> 5;
    int lane = tid & 31;
    int row  = blockIdx.x * 8 + w;

    const float* p = emb + (size_t)row * DIM + lane * 8;
    float4 v0 = *reinterpret_cast<const float4*>(p);
    float4 v1 = *reinterpret_cast<const float4*>(p + 4);
    float s = v0.x*v0.x + v0.y*v0.y + v0.z*v0.z + v0.w*v0.w
            + v1.x*v1.x + v1.y*v1.y + v1.z*v1.z + v1.w*v1.w;
    #pragma unroll
    for (int m = 1; m < 32; m <<= 1) s += __shfl_xor(s, m, 32);

    float inv = 1.0f / fmaxf(sqrtf(s), 1e-12f);

    alignas(16) _Float16 o[8];
    o[0] = (_Float16)(v0.x * inv); o[1] = (_Float16)(v0.y * inv);
    o[2] = (_Float16)(v0.z * inv); o[3] = (_Float16)(v0.w * inv);
    o[4] = (_Float16)(v1.x * inv); o[5] = (_Float16)(v1.y * inv);
    o[6] = (_Float16)(v1.z * inv); o[7] = (_Float16)(v1.w * inv);
    *reinterpret_cast<uint4*>(out + (size_t)row * DIM + lane * 8) =
        *reinterpret_cast<const uint4*>(o);
}

// Fused sim-GEMM + row-wise contrastive statistics.
// Grid: BATCH/128 blocks of 256 threads (8 waves). Wave w owns rows
// [blk*128 + w*16, +16). Sweeps all BATCH columns in 16-wide chunks.
// State per row: sumE (Σ exp, excl. self), posS (Σ exp over positives),
// posM (max positive sim), t1>=t2>=t3 (top-3 negative sims; t1 == max_neg).
__global__ __launch_bounds__(256) void sim_stats_kernel(
        const _Float16* __restrict__ ef, const int* __restrict__ labels,
        float* __restrict__ acc) {
    __shared__ float red[8][5];

    int tid   = threadIdx.x;
    int w     = tid >> 5;
    int lane  = tid & 31;
    int laneN = lane & 15;   // column within tile / row index for A
    int hi    = lane >> 4;   // lane half selector
    int rowTile = blockIdx.x * 128 + w * 16;

    // Wave-uniform copy of rowTile -> scalar diag-chunk index.
    int rowTileS   = __builtin_amdgcn_readfirstlane(rowTile);
    const int diagIt = rowTileS >> 4;   // self hits only in chunk it == diagIt

    // Loop-invariant per-lane masks: does C element r belong to the diagonal
    // position of its tile? (row-in-tile hi*8+r == col-in-tile laneN)
    bool selfLane[8];
    #pragma unroll
    for (int r = 0; r < 8; ++r) selfLane[r] = (laneN == hi * 8 + r);

    // ---- A fragments (16x32 f16, per ISA lane layout), K = 256 = 8 steps ----
    v16h afrag[8];
    {
        const _Float16* ap = ef + (size_t)(rowTile + laneN) * DIM;
        #pragma unroll
        for (int ks = 0; ks < 8; ++ks) {
            int off = ks * 32 + hi * 8;             // lanes 16-31 hold K=8..15,24..31
            afrag[ks] = load_frag(ap + off, ap + off + 16);
        }
    }

    // Labels for this lane's 8 rows (rows r + 8*hi in C layout).
    int rlab[8];
    {
        const int* lp = labels + rowTile + hi * 8;
        #pragma unroll
        for (int r = 0; r < 8; ++r) rlab[r] = lp[r];
    }

    float sumE[8], posS[8], posM[8], t1[8], t2[8], t3[8];
    #pragma unroll
    for (int r = 0; r < 8; ++r) {
        sumE[r] = 0.0f; posS[r] = 0.0f;
        posM[r] = NEG_BIG;
        t1[r] = NEG_BIG; t2[r] = NEG_BIG; t3[r] = NEG_BIG;
    }

    const int NCHUNK = BATCH / 16;
    #pragma unroll 2
    for (int it = 0; it < NCHUNK; ++it) {
        int colBase = it * 16;
        int bcol    = colBase + laneN;
        const _Float16* bp = ef + (size_t)bcol * DIM + hi * 8;  // B 32x16 layout
        __builtin_prefetch(ef + (size_t)(bcol + 16) * DIM + hi * 16, 0, 3);
        int clab = labels[bcol];
        bool diag = (it == diagIt);   // wave-uniform (scalar) condition

        v8f c = {};
        #pragma unroll
        for (int ks = 0; ks < 8; ++ks) {
            v16h bfrag = load_frag(bp + ks * 32, bp + ks * 32 + 16);
            c = __builtin_amdgcn_wmma_f32_16x16x32_f16(
                    false, afrag[ks], false, bfrag, (short)0, c, false, false);
        }

        // Branch-free epilogue: c[r] = cos(row rowTile+hi*8+r, col bcol)
        #pragma unroll
        for (int r = 0; r < 8; ++r) {
            float s    = c[r] * INV_T;
            bool  self = diag && selfLane[r];        // scalar AND lane-mask: SALU
            bool  same = (rlab[r] == clab);
            bool  pos  = same && !self;
            float e    = __expf(s);

            sumE[r] += self ? 0.0f : e;
            posS[r] += pos  ? e    : 0.0f;
            posM[r]  = fmaxf(posM[r], pos ? s : NEG_BIG);

            // branchless sorted insert of this negative into (t1>=t2>=t3)
            float ns = same ? NEG_BIG : s;
            float m1 = fmaxf(t1[r], ns);
            float r1 = fminf(t1[r], ns);
            float m2 = fmaxf(t2[r], r1);
            float r2 = fminf(t2[r], r1);
            t3[r] = fmaxf(t3[r], r2);
            t1[r] = m1;
            t2[r] = m2;
        }
    }

    // ---- reduce across the 16 lanes that share each row (branchless merge) ----
    #pragma unroll
    for (int m = 1; m < 16; m <<= 1) {
        #pragma unroll
        for (int r = 0; r < 8; ++r) {
            sumE[r] += __shfl_xor(sumE[r], m, 32);
            posS[r] += __shfl_xor(posS[r], m, 32);
            posM[r]  = fmaxf(posM[r], __shfl_xor(posM[r], m, 32));
            float o1 = __shfl_xor(t1[r], m, 32);
            float o2 = __shfl_xor(t2[r], m, 32);
            float o3 = __shfl_xor(t3[r], m, 32);
            #pragma unroll
            for (int q = 0; q < 3; ++q) {
                float s  = (q == 0) ? o1 : (q == 1) ? o2 : o3;
                float m1 = fmaxf(t1[r], s);
                float r1 = fminf(t1[r], s);
                float m2 = fmaxf(t2[r], r1);
                float r2 = fminf(t2[r], r1);
                t3[r] = fmaxf(t3[r], r2);
                t1[r] = m1;
                t2[r] = m2;
            }
        }
    }

    // Per-lane partial loss sums over this half's 8 rows.
    float aB = 0.0f, aHp = 0.0f, aH = 0.0f, aM = 0.0f, aV = 0.0f;
    #pragma unroll
    for (int r = 0; r < 8; ++r) {
        bool hasPos = posM[r] > NEG_BIG * 0.5f;
        bool hasNeg = t1[r]   > NEG_BIG * 0.5f;   // t1 == max negative
        float basic = -logf(posS[r] / (sumE[r] + 1e-10f) + 1e-10f);
        float hn    = (t1[r] + t2[r] + t3[r]) * (1.0f / 3.0f);
        float h     = fmaxf(hn - posM[r] + MARGIN_V, 0.0f);
        float mg    = fmaxf(t1[r] - posM[r] + MARGIN_V, 0.0f);
        aB  += hasPos ? basic : 0.0f;
        aHp += hasPos ? 1.0f  : 0.0f;
        aH  += hasPos ? h     : 0.0f;
        aM  += (hasPos && hasNeg) ? mg   : 0.0f;
        aV  += (hasPos && hasNeg) ? 1.0f : 0.0f;
    }
    // combine the two 16-lane halves
    aB  += __shfl_xor(aB, 16, 32);
    aHp += __shfl_xor(aHp, 16, 32);
    aH  += __shfl_xor(aH, 16, 32);
    aM  += __shfl_xor(aM, 16, 32);
    aV  += __shfl_xor(aV, 16, 32);

    if (lane == 0) {
        red[w][0] = aB; red[w][1] = aHp; red[w][2] = aH;
        red[w][3] = aM; red[w][4] = aV;
    }
    __syncthreads();
    if (tid < 5) {
        float s = 0.0f;
        #pragma unroll
        for (int ww = 0; ww < 8; ++ww) s += red[ww][tid];
        atomicAdd(&acc[tid], s);
    }
}

__global__ void finalize_kernel(const float* __restrict__ acc,
                                float* __restrict__ out) {
    if (threadIdx.x == 0 && blockIdx.x == 0) {
        float nhp    = fmaxf(acc[1], 1.0f);
        float basic  = acc[0] / nhp;
        float hard   = acc[2] / nhp;
        float nv     = fmaxf(acc[4], 1.0f);
        float margin = (acc[4] > 0.0f) ? (acc[3] / nv) : 0.0f;
        out[0] = basic + 0.5f * hard + 0.1f * margin;
    }
}

extern "C" void kernel_launch(void* const* d_in, const int* in_sizes, int n_in,
                              void* d_out, int out_size, void* d_ws, size_t ws_size,
                              hipStream_t stream) {
    const float* emb    = (const float*)d_in[0];
    const int*   labels = (const int*)d_in[1];
    float*       out    = (float*)d_out;

    // Workspace layout: [0,32): accumulators; [256, 256+4MB): f16 matrix.
    float*     acc = (float*)d_ws;
    _Float16*  ef  = (_Float16*)((char*)d_ws + 256);

    zero_acc_kernel<<<1, 32, 0, stream>>>(acc);
    normalize_f16_kernel<<<BATCH / 8, 256, 0, stream>>>(emb, ef);
    sim_stats_kernel<<<BATCH / 128, 256, 0, stream>>>(ef, labels, acc);
    finalize_kernel<<<1, 32, 0, stream>>>(acc, out);
}
